// TransMIL_74105365725539
// MI455X (gfx1250) — compile-verified, matrix operand order checked by reading, not run on hardware
//
#include <hip/hip_runtime.h>
#include <hip/hip_bf16.h>
#include <stdint.h>

typedef _Float16 half_t;
typedef __attribute__((ext_vector_type(16))) _Float16 v16h;
typedef __attribute__((ext_vector_type(8)))  float    v8f;

#define NTOK   6000
#define INDIM  1536
#define DIMC   1024
#define EODIM  3072
#define HEADS_ 8
#define DHEAD_ 128
#define LMK_   512
#define NP_    6144
#define PADF_  59
#define TOKS_  6085
#define FEAT_  6084
#define HW_    78
#define LNUM_  12
#define QKVN   3072
#define QSCALE 0.08838834764831845f

static inline int cdiv_i(long a, int b){ return (int)((a + b - 1)/b); }

// ---------------- block reductions (wave32) ----------------
__device__ __forceinline__ float blockReduceSum(float v, float* sh) {
  __syncthreads();
  #pragma unroll
  for (int o = 16; o; o >>= 1) v += __shfl_down(v, o, 32);
  int wid = threadIdx.x >> 5;
  if ((threadIdx.x & 31) == 0) sh[wid] = v;
  __syncthreads();
  int nw = blockDim.x >> 5;
  v = (threadIdx.x < nw) ? sh[threadIdx.x] : 0.f;
  if (wid == 0) {
    #pragma unroll
    for (int o = 4; o; o >>= 1) v += __shfl_down(v, o, 32);
    if (threadIdx.x == 0) sh[0] = v;
  }
  __syncthreads();
  return sh[0];
}

__device__ __forceinline__ float blockReduceMax(float v, float* sh) {
  __syncthreads();
  #pragma unroll
  for (int o = 16; o; o >>= 1) v = fmaxf(v, __shfl_down(v, o, 32));
  int wid = threadIdx.x >> 5;
  if ((threadIdx.x & 31) == 0) sh[wid] = v;
  __syncthreads();
  int nw = blockDim.x >> 5;
  v = (threadIdx.x < nw) ? sh[threadIdx.x] : -3.4e38f;
  if (wid == 0) {
    #pragma unroll
    for (int o = 4; o; o >>= 1) v = fmaxf(v, __shfl_down(v, o, 32));
    if (threadIdx.x == 0) sh[0] = v;
  }
  __syncthreads();
  return sh[0];
}

// ---------------- WMMA GEMM: C[M,N] = A[M,K] (f16) x B (f16), f32 accum ----------------
// TRANSB=true : B stored (N,K) row-major (weight-style, contiguous K) -> direct
//               clause-pipelined global_load_b128 + prefetch path.
// TRANSB=false: B stored (K,N) row-major -> 32x128 B tile staged in LDS with
//               global_load_async_to_lds_b128 (one b128 per thread = 4KB of the 8KB tile),
//               s_wait_asynccnt + barrier, fragments gathered from LDS.
// Block = 256 threads = 8 waves, each wave one 16x16 tile; block tile 16x128.
// All call sites have M%16==0, N%128==0, K%32==0.
template<bool TRANSB>
__global__ __launch_bounds__(256)
void k_gemm_wmma(const half_t* __restrict__ A, const half_t* __restrict__ B,
                 float* __restrict__ C, int M, int N, int K,
                 long sA, long sB, long sC)
{
  const int wave = threadIdx.x >> 5;
  const int lane = threadIdx.x & 31;
  const long batch = blockIdx.z;
  A += batch * sA; B += batch * sB; C += batch * sC;
  const int m0   = blockIdx.y << 4;
  const int nblk = blockIdx.x << 7;
  const int n0   = nblk + (wave << 4);
  if (m0 >= M || n0 >= N) return;          // uniform across block (N%128==0)
  const int ar  = m0 + (lane & 15);
  const int ks0 = (lane >> 4) << 4;        // 0 or 16
  const int bn  = n0 + (lane & 15);
  const half_t* Arow = A + (long)ar * K;
  v8f acc = {};

  if constexpr (TRANSB) {
    for (int k0 = 0; k0 < K; k0 += 32) {
      v16h a = *(const v16h*)(Arow + k0 + ks0);
      v16h b = *(const v16h*)(B + (long)bn * K + k0 + ks0);
      if (k0 + 32 < K) __builtin_prefetch(Arow + k0 + 32 + ks0, 0, 3);
      acc = __builtin_amdgcn_wmma_f32_16x16x32_f16(false, a, false, b,
                                                   (short)0, acc, false, false);
    }
  } else {
    __shared__ half_t Btile[32 * 128];     // 8 KB; sole shared object -> LDS offset 0
    const int tid  = threadIdx.x;
    const int brow = tid >> 3;             // 0..31  (K within tile)
    const int bcol = (tid & 7) << 4;       // 0,16,...,112 (N within tile)
    const unsigned lds_off = (unsigned)tid * 32u;   // == (brow*128 + bcol)*2 bytes
    const int bfc = (wave << 4) + (lane & 15);      // fragment column within tile
    for (int k0 = 0; k0 < K; k0 += 32) {
      const half_t* gp = B + (long)(k0 + brow) * N + nblk + bcol;
      // async coalesced stage of the 32x128 B tile into LDS (ASYNCcnt path)
      asm volatile("global_load_async_to_lds_b128 %0, %1, off"
                   :: "v"(lds_off), "v"(gp) : "memory");
      v16h a = *(const v16h*)(Arow + k0 + ks0);
      asm volatile("s_wait_asynccnt 0x0" ::: "memory");
      __syncthreads();
      v16h b;
      #pragma unroll
      for (int j = 0; j < 16; ++j) b[j] = Btile[(ks0 + j) * 128 + bfc];
      acc = __builtin_amdgcn_wmma_f32_16x16x32_f16(false, a, false, b,
                                                   (short)0, acc, false, false);
      __syncthreads();                     // protect tile before next stage
    }
  }

  // C layout: VGPR r -> row m0 + r + (lane>>4)*8, col n0 + (lane&15)
  const int crow = m0 + ((lane >> 4) << 3);
  float* Crow = C + (long)crow * N + bn;
  #pragma unroll
  for (int r = 0; r < 8; ++r) Crow[(long)r * N] = acc[r];
}

template<bool TB>
static void gemm(const half_t* A, const half_t* B, float* C, int M, int N, int K,
                 int batch, long sA, long sB, long sC, hipStream_t st){
  dim3 g((unsigned)(N/128), (unsigned)(M/16), (unsigned)batch);
  k_gemm_wmma<TB><<<g, 256, 0, st>>>(A, B, C, M, N, K, sA, sB, sC);
}

// ---------------- elementwise / utility kernels ----------------
__global__ __launch_bounds__(256)
void k_f32_to_f16(const float* __restrict__ s, half_t* __restrict__ d, long n){
  long i = (long)blockIdx.x * blockDim.x + threadIdx.x;
  if (i < n) d[i] = (half_t)s[i];
}

__global__ __launch_bounds__(256)
void k_copy_f32(const float* __restrict__ s, float* __restrict__ d, long n){
  long i = (long)blockIdx.x * blockDim.x + threadIdx.x;
  if (i < n) d[i] = s[i];
}

__global__ __launch_bounds__(256)
void k_gates(const float* __restrict__ X, const float* __restrict__ wg,
             float* __restrict__ gates){
  int t = blockIdx.x;
  const float* x = X + (long)t * INDIM;
  float a0 = 0.f, a1 = 0.f, a2 = 0.f;
  for (int f = threadIdx.x; f < INDIM; f += blockDim.x) {
    float xv = x[f];
    a0 += xv * wg[f*3+0]; a1 += xv * wg[f*3+1]; a2 += xv * wg[f*3+2];
  }
  __shared__ float sh[8];
  a0 = blockReduceSum(a0, sh);
  a1 = blockReduceSum(a1, sh);
  a2 = blockReduceSum(a2, sh);
  if (threadIdx.x == 0) {
    float m = fmaxf(a0, fmaxf(a1, a2));
    float e0 = __expf(a0-m), e1 = __expf(a1-m), e2 = __expf(a2-m);
    float s = e0 + e1 + e2;
    gates[t*3+0] = e0/s; gates[t*3+1] = e1/s; gates[t*3+2] = e2/s;
  }
}

__global__ __launch_bounds__(256)
void k_moe_combine(const float* __restrict__ Eo, const float* __restrict__ gates,
                   const float* __restrict__ eb, const float* __restrict__ cls,
                   float* __restrict__ Hb){
  int r = blockIdx.x;                 // 0..TOKS_-1
  float* o = Hb + (long)r * DIMC;
  if (r == 0) { for (int c = threadIdx.x; c < DIMC; c += blockDim.x) o[c] = cls[c]; return; }
  int i = r - 1;
  int s = (i < NTOK) ? i : (i - NTOK);
  const float* eo = Eo + (long)s * EODIM;
  float g0 = gates[s*3], g1 = gates[s*3+1], g2 = gates[s*3+2];
  for (int c = threadIdx.x; c < DIMC; c += blockDim.x) {
    float v0 = fmaxf(eo[c]            + eb[c],           0.f);
    float v1 = fmaxf(eo[DIMC + c]     + eb[DIMC + c],    0.f);
    float v2 = fmaxf(eo[2*DIMC + c]   + eb[2*DIMC + c],  0.f);
    o[c] = g0*v0 + g1*v1 + g2*v2;
  }
}

__global__ __launch_bounds__(256)
void k_ln_to_f16(const float* __restrict__ Hb, half_t* __restrict__ Xp,
                 const float* __restrict__ g, const float* __restrict__ b){
  int r = blockIdx.x;                 // 0..NP_-1
  half_t* o = Xp + (long)r * DIMC;
  if (r < PADF_) {
    for (int c = threadIdx.x; c < DIMC; c += blockDim.x) o[c] = (half_t)0.f;
    return;
  }
  const float* x = Hb + (long)(r - PADF_) * DIMC;
  __shared__ float sh[8];
  float s = 0.f, s2 = 0.f;
  for (int c = threadIdx.x; c < DIMC; c += blockDim.x) { float v = x[c]; s += v; s2 += v*v; }
  s  = blockReduceSum(s,  sh);
  s2 = blockReduceSum(s2, sh);
  float mu = s / (float)DIMC;
  float rs = rsqrtf(s2 / (float)DIMC - mu*mu + 1e-5f);
  for (int c = threadIdx.x; c < DIMC; c += blockDim.x)
    o[c] = (half_t)((x[c] - mu) * rs * g[c] + b[c]);
}

__global__ __launch_bounds__(256)
void k_split_qkv(const float* __restrict__ QKV, half_t* __restrict__ Qh,
                 half_t* __restrict__ Kh, half_t* __restrict__ Vh){
  long idx = (long)blockIdx.x * blockDim.x + threadIdx.x;
  if (idx >= (long)NP_ * QKVN) return;
  int col = (int)(idx % QKVN);
  long row = idx / QKVN;
  int which = col >> 10;
  int head  = (col & 1023) >> 7;
  int d     = col & 127;
  float v = QKV[idx];
  long dst = ((long)head * NP_ + row) * DHEAD_ + d;
  if      (which == 0) Qh[dst] = (half_t)(v * QSCALE);
  else if (which == 1) Kh[dst] = (half_t)v;
  else                 Vh[dst] = (half_t)v;
}

__global__ __launch_bounds__(128)
void k_landmarks(const half_t* __restrict__ Qh, const half_t* __restrict__ Kh,
                 half_t* __restrict__ QLh, half_t* __restrict__ KLh){
  int m = blockIdx.x, h = blockIdx.y, d = threadIdx.x;
  float sq = 0.f, sk = 0.f;
  long base = ((long)h * NP_ + (long)m * LNUM_) * DHEAD_ + d;
  #pragma unroll
  for (int j = 0; j < LNUM_; ++j) {
    sq += (float)Qh[base + (long)j * DHEAD_];
    sk += (float)Kh[base + (long)j * DHEAD_];
  }
  long o = ((long)h * LMK_ + m) * DHEAD_ + d;
  QLh[o] = (half_t)(sq / (float)LNUM_);
  KLh[o] = (half_t)(sk / (float)LNUM_);
}

__global__ __launch_bounds__(256)
void k_softmax_rows(const float* __restrict__ S, half_t* __restrict__ O, int cols){
  long row = blockIdx.x;
  const float* s = S + row * (long)cols;
  half_t* o = O + row * (long)cols;
  __shared__ float sh[8];
  float mx = -3.4e38f;
  for (int c = threadIdx.x; c < cols; c += blockDim.x) mx = fmaxf(mx, s[c]);
  mx = blockReduceMax(mx, sh);
  float sum = 0.f;
  for (int c = threadIdx.x; c < cols; c += blockDim.x) sum += __expf(s[c] - mx);
  sum = blockReduceSum(sum, sh);
  float inv = 1.f / sum;
  for (int c = threadIdx.x; c < cols; c += blockDim.x)
    o[c] = (half_t)(__expf(s[c] - mx) * inv);
}

__global__ void k_init_scal(float* scal){ if (threadIdx.x < 2) scal[threadIdx.x] = 0.f; }

__device__ __forceinline__ void atomicMaxPosF(float* a, float v){
  atomicMax((unsigned int*)a, __float_as_uint(v));   // valid for non-negative floats
}

__global__ __launch_bounds__(256)
void k_rowmax(const half_t* __restrict__ A2h, float* __restrict__ scal){
  int i = blockIdx.x, h = blockIdx.y;
  const half_t* row = A2h + ((long)h * LMK_ + i) * LMK_;
  float s = 0.f;
  for (int j = threadIdx.x; j < LMK_; j += blockDim.x) s += fabsf((float)row[j]);
  __shared__ float sh[8];
  s = blockReduceSum(s, sh);
  if (threadIdx.x == 0) atomicMaxPosF(scal + 0, s);
}

__global__ __launch_bounds__(256)
void k_colmax(const half_t* __restrict__ A2h, float* __restrict__ scal){
  int j = blockIdx.x, h = blockIdx.y;
  const half_t* base = A2h + (long)h * LMK_ * LMK_ + j;
  float s = 0.f;
  for (int i = threadIdx.x; i < LMK_; i += blockDim.x) s += fabsf((float)base[(long)i * LMK_]);
  __shared__ float sh[8];
  s = blockReduceSum(s, sh);
  if (threadIdx.x == 0) atomicMaxPosF(scal + 1, s);
}

__global__ __launch_bounds__(256)
void k_zinit(const half_t* __restrict__ A2h, half_t* __restrict__ Zh,
             const float* __restrict__ scal){
  long idx = (long)blockIdx.x * blockDim.x + threadIdx.x;
  if (idx >= (long)HEADS_ * LMK_ * LMK_) return;
  float inv = 1.f / (scal[0] * scal[1]);
  long h = idx / ((long)LMK_ * LMK_);
  long rem = idx - h * (long)LMK_ * LMK_;
  long i = rem / LMK_, j = rem - i * LMK_;
  // z = x^T / (maxrow*maxcol)
  Zh[idx] = (half_t)((float)A2h[(h * LMK_ + j) * LMK_ + i] * inv);
}

__global__ __launch_bounds__(256)
void k_aI_minus(const float* __restrict__ X, half_t* __restrict__ O, float alpha){
  long idx = (long)blockIdx.x * blockDim.x + threadIdx.x;
  if (idx >= (long)HEADS_ * LMK_ * LMK_) return;
  long rem = idx % ((long)LMK_ * LMK_);
  long i = rem / LMK_, j = rem - i * LMK_;
  float d = (i == j) ? alpha : 0.f;
  O[idx] = (half_t)(d - X[idx]);
}

__global__ __launch_bounds__(256)
void k_scale025_f16(const float* __restrict__ X, half_t* __restrict__ O, long n){
  long i = (long)blockIdx.x * blockDim.x + threadIdx.x;
  if (i < n) O[i] = (half_t)(0.25f * X[i]);
}

__global__ __launch_bounds__(128)
void k_res_conv_add(const half_t* __restrict__ Vh, const float* __restrict__ rw,
                    float* __restrict__ OutAttn){
  int d = threadIdx.x;     // 0..127
  int n = blockIdx.x;      // 0..NP_-1
  int h = blockIdx.y;      // 0..7
  float acc = 0.f;
  #pragma unroll
  for (int t = 0; t < 33; ++t) {
    int src = n + t - 16;
    if (src >= 0 && src < NP_)
      acc += (float)Vh[((long)h * NP_ + src) * DHEAD_ + d] * rw[h * 33 + t];
  }
  OutAttn[((long)h * NP_ + n) * DHEAD_ + d] += acc;
}

__global__ __launch_bounds__(256)
void k_concat_heads(const float* __restrict__ OutAttn, half_t* __restrict__ OutCatH){
  long idx = (long)blockIdx.x * blockDim.x + threadIdx.x;
  if (idx >= (long)NP_ * DIMC) return;
  int c = (int)(idx & (DIMC - 1));
  long n = idx >> 10;
  int h = c >> 7, d = c & 127;
  OutCatH[idx] = (half_t)OutAttn[((long)h * NP_ + n) * DHEAD_ + d];
}

__global__ __launch_bounds__(256)
void k_resid_add(float* __restrict__ Hb, const float* __restrict__ Proj,
                 const float* __restrict__ ob){
  long idx = (long)blockIdx.x * blockDim.x + threadIdx.x;
  if (idx >= (long)TOKS_ * DIMC) return;
  int c = (int)(idx & (DIMC - 1));
  long t = idx >> 10;
  Hb[idx] += Proj[(long)(PADF_ + t) * DIMC + c] + ob[c];
}

__global__ __launch_bounds__(256)
void k_ppeg(const float* __restrict__ Hin, float* __restrict__ Hout,
            const float* __restrict__ pw7, const float* __restrict__ pb7,
            const float* __restrict__ pw5, const float* __restrict__ pb5,
            const float* __restrict__ pw3, const float* __restrict__ pb3){
  long idx = (long)blockIdx.x * blockDim.x + threadIdx.x;
  if (idx >= (long)DIMC * FEAT_) return;
  int c = (int)(idx / FEAT_);
  int p = (int)(idx - (long)c * FEAT_);
  int y = p / HW_, x = p - y * HW_;
  float acc = Hin[(long)(1 + p) * DIMC + c] + pb7[c] + pb5[c] + pb3[c];
  const float* w7 = pw7 + c * 49;
  const float* w5 = pw5 + c * 25;
  const float* w3 = pw3 + c * 9;
  for (int dy = -3; dy <= 3; ++dy) {
    int yy = y + dy; if ((unsigned)yy >= (unsigned)HW_) continue;
    for (int dx = -3; dx <= 3; ++dx) {
      int xx = x + dx; if ((unsigned)xx >= (unsigned)HW_) continue;
      float v = Hin[(long)(1 + yy * HW_ + xx) * DIMC + c];
      acc += v * w7[(dy + 3) * 7 + (dx + 3)];
      if (dy >= -2 && dy <= 2 && dx >= -2 && dx <= 2) acc += v * w5[(dy + 2) * 5 + (dx + 2)];
      if (dy >= -1 && dy <= 1 && dx >= -1 && dx <= 1) acc += v * w3[(dy + 1) * 3 + (dx + 1)];
    }
  }
  Hout[(long)(1 + p) * DIMC + c] = acc;
}

__global__ __launch_bounds__(256)
void k_head(const float* __restrict__ Hb, const float* __restrict__ g,
            const float* __restrict__ b, const float* __restrict__ fw,
            const float* __restrict__ fb, float* __restrict__ out){
  __shared__ float sh[8];
  const float* x = Hb;   // cls row
  float s = 0.f, s2 = 0.f;
  for (int c = threadIdx.x; c < DIMC; c += blockDim.x) { float v = x[c]; s += v; s2 += v*v; }
  s  = blockReduceSum(s,  sh);
  s2 = blockReduceSum(s2, sh);
  float mu = s / (float)DIMC;
  float rs = rsqrtf(s2 / (float)DIMC - mu*mu + 1e-5f);
  float d0 = 0.f, d1 = 0.f;
  for (int c = threadIdx.x; c < DIMC; c += blockDim.x) {
    float v = (x[c] - mu) * rs * g[c] + b[c];
    d0 += v * fw[c];
    d1 += v * fw[DIMC + c];
  }
  d0 = blockReduceSum(d0, sh);
  d1 = blockReduceSum(d1, sh);
  if (threadIdx.x == 0) {
    float l0 = d0 + fb[0], l1 = d1 + fb[1];
    out[0] = l0; out[1] = l1;
    float m = fmaxf(l0, l1);
    float e0 = __expf(l0 - m), e1 = __expf(l1 - m), sm = e0 + e1;
    out[2] = e0 / sm; out[3] = e1 / sm;
    out[4] = (l1 > l0) ? 1.f : 0.f;
  }
}

// ---------------- workspace carving ----------------
struct Bufs {
  float  *BIGA, *PINVF, *gates, *Hbuf, *Hbuf2, *scal;
  half_t *Xh, *WexpH, *XpH, *WqkvH, *WoutH, *Qh, *Kh, *Vh, *QLh, *KLh,
         *A1h, *A3h, *A2h, *Zh, *Ph, *Ta, *Tb, *A3Vh, *W2h, *OutCatH;
};

static Bufs carve(void* ws){
  char* p = (char*)ws;
  Bufs b;
  auto take = [&](size_t bytes)->char* {
    char* r = p;
    p += (bytes + 255) & ~(size_t)255;
    return r;
  };
  b.BIGA   = (float*) take((size_t)HEADS_ * NP_ * LMK_ * 4);        // 100.7 MB (Eo/QKV/S1/S3/OutAttn/Proj)
  b.PINVF  = (float*) take((size_t)HEADS_ * LMK_ * LMK_ * 4);       // pinv f32 scratch (+ A3V/W2 f32)
  b.gates  = (float*) take((size_t)NTOK * 3 * 4);
  b.Hbuf   = (float*) take((size_t)TOKS_ * DIMC * 4);
  b.Hbuf2  = (float*) take((size_t)TOKS_ * DIMC * 4);
  b.scal   = (float*) take(256);
  b.Xh     = (half_t*)take((size_t)NTOK * INDIM * 2);
  b.WexpH  = (half_t*)take((size_t)EODIM * INDIM * 2);
  b.XpH    = (half_t*)take((size_t)NP_ * DIMC * 2);
  b.WqkvH  = (half_t*)take((size_t)QKVN * DIMC * 2);
  b.WoutH  = (half_t*)take((size_t)DIMC * DIMC * 2);
  b.Qh     = (half_t*)take((size_t)HEADS_ * NP_ * DHEAD_ * 2);
  b.Kh     = (half_t*)take((size_t)HEADS_ * NP_ * DHEAD_ * 2);
  b.Vh     = (half_t*)take((size_t)HEADS_ * NP_ * DHEAD_ * 2);
  b.QLh    = (half_t*)take((size_t)HEADS_ * LMK_ * DHEAD_ * 2);
  b.KLh    = (half_t*)take((size_t)HEADS_ * LMK_ * DHEAD_ * 2);
  b.A1h    = (half_t*)take((size_t)HEADS_ * NP_ * LMK_ * 2);
  b.A3h    = (half_t*)take((size_t)HEADS_ * LMK_ * NP_ * 2);
  b.A2h    = (half_t*)take((size_t)HEADS_ * LMK_ * LMK_ * 2);
  b.Zh     = (half_t*)take((size_t)HEADS_ * LMK_ * LMK_ * 2);
  b.Ph     = (half_t*)take((size_t)HEADS_ * LMK_ * LMK_ * 2);
  b.Ta     = (half_t*)take((size_t)HEADS_ * LMK_ * LMK_ * 2);
  b.Tb     = (half_t*)take((size_t)HEADS_ * LMK_ * LMK_ * 2);
  b.A3Vh   = (half_t*)take((size_t)HEADS_ * LMK_ * DHEAD_ * 2);
  b.W2h    = (half_t*)take((size_t)HEADS_ * LMK_ * DHEAD_ * 2);
  b.OutCatH= (half_t*)take((size_t)NP_ * DIMC * 2);
  return b;
}

static inline void ew(long n, hipStream_t st, dim3& g){ (void)st; g = dim3((unsigned)cdiv_i(n, 256)); }

// ---------------- one Nystrom attention block (in-place on Hb) ----------------
static void run_nystrom(float* Hb,
                        const float* ln_g, const float* ln_b,
                        const float* qkv_w, const float* out_w, const float* out_b,
                        const float* res_w, const Bufs& B, hipStream_t st)
{
  dim3 g;
  const long NSQ = (long)HEADS_ * LMK_ * LMK_;

  // weights -> f16
  ew((long)QKVN * DIMC, st, g);
  k_f32_to_f16<<<g, 256, 0, st>>>(qkv_w, B.WqkvH, (long)QKVN * DIMC);
  ew((long)DIMC * DIMC, st, g);
  k_f32_to_f16<<<g, 256, 0, st>>>(out_w, B.WoutH, (long)DIMC * DIMC);

  // layernorm + front pad -> XpH (6144 x 1024 f16)
  k_ln_to_f16<<<dim3(NP_), 256, 0, st>>>(Hb, B.XpH, ln_g, ln_b);

  // QKV = XpH @ qkv_w^T   (6144 x 3072)
  gemm<true>(B.XpH, B.WqkvH, B.BIGA, NP_, QKVN, DIMC, 1, 0, 0, 0, st);

  // split into per-head Q (scaled), K, V f16
  ew((long)NP_ * QKVN, st, g);
  k_split_qkv<<<g, 256, 0, st>>>(B.BIGA, B.Qh, B.Kh, B.Vh);

  // landmarks
  k_landmarks<<<dim3(LMK_, HEADS_), 128, 0, st>>>(B.Qh, B.Kh, B.QLh, B.KLh);

  // a1 = softmax(q @ kl^T)   (8 x 6144 x 512)
  gemm<true>(B.Qh, B.KLh, B.BIGA, NP_, LMK_, DHEAD_, HEADS_,
             (long)NP_*DHEAD_, (long)LMK_*DHEAD_, (long)NP_*LMK_, st);
  k_softmax_rows<<<dim3(HEADS_*NP_), 256, 0, st>>>(B.BIGA, B.A1h, LMK_);

  // a2 = softmax(ql @ kl^T)  (8 x 512 x 512)
  gemm<true>(B.QLh, B.KLh, B.PINVF, LMK_, LMK_, DHEAD_, HEADS_,
             (long)LMK_*DHEAD_, (long)LMK_*DHEAD_, (long)LMK_*LMK_, st);
  k_softmax_rows<<<dim3(HEADS_*LMK_), 256, 0, st>>>(B.PINVF, B.A2h, LMK_);

  // a3 = softmax(ql @ k^T)   (8 x 512 x 6144)
  gemm<true>(B.QLh, B.Kh, B.BIGA, LMK_, NP_, DHEAD_, HEADS_,
             (long)LMK_*DHEAD_, (long)NP_*DHEAD_, (long)LMK_*NP_, st);
  k_softmax_rows<<<dim3(HEADS_*LMK_), 256, 0, st>>>(B.BIGA, B.A3h, NP_);

  // ---- pinv(a2) via 6 Newton-Schulz iterations ----
  k_init_scal<<<dim3(1), 32, 0, st>>>(B.scal);
  k_rowmax<<<dim3(LMK_, HEADS_), 256, 0, st>>>(B.A2h, B.scal);
  k_colmax<<<dim3(LMK_, HEADS_), 256, 0, st>>>(B.A2h, B.scal);
  ew(NSQ, st, g);
  k_zinit<<<g, 256, 0, st>>>(B.A2h, B.Zh, B.scal);
  for (int it = 0; it < 6; ++it) {
    // P = a2 @ z
    gemm<false>(B.A2h, B.Zh, B.PINVF, LMK_, LMK_, LMK_, HEADS_,
                (long)LMK_*LMK_, (long)LMK_*LMK_, (long)LMK_*LMK_, st);
    ew(NSQ, st, g);
    k_f32_to_f16<<<g, 256, 0, st>>>(B.PINVF, B.Ph, NSQ);
    k_aI_minus<<<g, 256, 0, st>>>(B.PINVF, B.Ta, 7.0f);
    // T2 = 15I - P @ Ta
    gemm<false>(B.Ph, B.Ta, B.PINVF, LMK_, LMK_, LMK_, HEADS_,
                (long)LMK_*LMK_, (long)LMK_*LMK_, (long)LMK_*LMK_, st);
    k_aI_minus<<<g, 256, 0, st>>>(B.PINVF, B.Tb, 15.0f);
    // T3 = 13I - P @ Tb
    gemm<false>(B.Ph, B.Tb, B.PINVF, LMK_, LMK_, LMK_, HEADS_,
                (long)LMK_*LMK_, (long)LMK_*LMK_, (long)LMK_*LMK_, st);
    k_aI_minus<<<g, 256, 0, st>>>(B.PINVF, B.Ta, 13.0f);
    // z = 0.25 * z @ T3
    gemm<false>(B.Zh, B.Ta, B.PINVF, LMK_, LMK_, LMK_, HEADS_,
                (long)LMK_*LMK_, (long)LMK_*LMK_, (long)LMK_*LMK_, st);
    k_scale025_f16<<<g, 256, 0, st>>>(B.PINVF, B.Zh, NSQ);
  }

  // a3v = a3 @ v   (8 x 512 x 128)
  gemm<false>(B.A3h, B.Vh, B.PINVF, LMK_, DHEAD_, NP_, HEADS_,
              (long)LMK_*NP_, (long)NP_*DHEAD_, (long)LMK_*DHEAD_, st);
  ew((long)HEADS_*LMK_*DHEAD_, st, g);
  k_f32_to_f16<<<g, 256, 0, st>>>(B.PINVF, B.A3Vh, (long)HEADS_*LMK_*DHEAD_);

  // w2 = pinv(a2) @ a3v
  gemm<false>(B.Zh, B.A3Vh, B.PINVF, LMK_, DHEAD_, LMK_, HEADS_,
              (long)LMK_*LMK_, (long)LMK_*DHEAD_, (long)LMK_*DHEAD_, st);
  k_f32_to_f16<<<g, 256, 0, st>>>(B.PINVF, B.W2h, (long)HEADS_*LMK_*DHEAD_);

  // out = a1 @ w2   (8 x 6144 x 128)  -> BIGA
  gemm<false>(B.A1h, B.W2h, B.BIGA, NP_, DHEAD_, LMK_, HEADS_,
              (long)NP_*LMK_, (long)LMK_*DHEAD_, (long)NP_*DHEAD_, st);

  // += depthwise 33-tap residual conv over v
  k_res_conv_add<<<dim3(NP_, HEADS_), 128, 0, st>>>(B.Vh, res_w, B.BIGA);

  // concat heads -> (6144 x 1024) f16
  ew((long)NP_ * DIMC, st, g);
  k_concat_heads<<<g, 256, 0, st>>>(B.BIGA, B.OutCatH);

  // proj = outcat @ out_w^T  -> BIGA (reuse)
  gemm<true>(B.OutCatH, B.WoutH, B.BIGA, NP_, DIMC, DIMC, 1, 0, 0, 0, st);

  // Hb += proj[last 6085 rows] + out_b
  ew((long)TOKS_ * DIMC, st, g);
  k_resid_add<<<g, 256, 0, st>>>(Hb, B.BIGA, out_b);
}

// ---------------- entry point ----------------
extern "C" void kernel_launch(void* const* d_in, const int* in_sizes, int n_in,
                              void* d_out, int out_size, void* d_ws, size_t ws_size,
                              hipStream_t stream) {
  (void)in_sizes; (void)n_in; (void)out_size; (void)ws_size;
  const float* data    = (const float*)d_in[0];
  const float* w_gate  = (const float*)d_in[1];
  const float* exp_w   = (const float*)d_in[2];
  const float* exp_b   = (const float*)d_in[3];
  const float* cls     = (const float*)d_in[4];
  const float* ln1_g   = (const float*)d_in[5];
  const float* ln1_b   = (const float*)d_in[6];
  const float* qkv1_w  = (const float*)d_in[7];
  const float* out1_w  = (const float*)d_in[8];
  const float* out1_b  = (const float*)d_in[9];
  const float* res1_w  = (const float*)d_in[10];
  const float* ln2_g   = (const float*)d_in[11];
  const float* ln2_b   = (const float*)d_in[12];
  const float* qkv2_w  = (const float*)d_in[13];
  const float* out2_w  = (const float*)d_in[14];
  const float* out2_b  = (const float*)d_in[15];
  const float* res2_w  = (const float*)d_in[16];
  const float* pw7     = (const float*)d_in[17];
  const float* pb7     = (const float*)d_in[18];
  const float* pw5     = (const float*)d_in[19];
  const float* pb5     = (const float*)d_in[20];
  const float* pw3     = (const float*)d_in[21];
  const float* pb3     = (const float*)d_in[22];
  const float* normf_g = (const float*)d_in[23];
  const float* normf_b = (const float*)d_in[24];
  const float* fc2_w   = (const float*)d_in[25];
  const float* fc2_b   = (const float*)d_in[26];
  float* out = (float*)d_out;

  Bufs B = carve(d_ws);
  dim3 g;

  // ---- MoE ----
  ew((long)NTOK * INDIM, stream, g);
  k_f32_to_f16<<<g, 256, 0, stream>>>(data, B.Xh, (long)NTOK * INDIM);
  ew((long)EODIM * INDIM, stream, g);
  k_f32_to_f16<<<g, 256, 0, stream>>>(exp_w, B.WexpH, (long)EODIM * INDIM);
  k_gates<<<dim3(NTOK), 256, 0, stream>>>(data, w_gate, B.gates);
  // Eo = X @ expert_w^T  (6000 x 3072)
  gemm<true>(B.Xh, B.WexpH, B.BIGA, NTOK, EODIM, INDIM, 1, 0, 0, 0, stream);
  // combine experts + square-pad tokens + cls -> Hbuf (6085 x 1024)
  k_moe_combine<<<dim3(TOKS_), 256, 0, stream>>>(B.BIGA, B.gates, exp_b, cls, B.Hbuf);

  // ---- block 1: h = h + nystrom(ln(h)) ----
  run_nystrom(B.Hbuf, ln1_g, ln1_b, qkv1_w, out1_w, out1_b, res1_w, B, stream);

  // ---- PPEG ----
  ew((long)DIMC * FEAT_, stream, g);
  k_ppeg<<<g, 256, 0, stream>>>(B.Hbuf, B.Hbuf2, pw7, pb7, pw5, pb5, pw3, pb3);
  k_copy_f32<<<dim3(4), 256, 0, stream>>>(B.Hbuf, B.Hbuf2, DIMC);  // cls row

  // ---- block 2 ----
  run_nystrom(B.Hbuf2, ln2_g, ln2_b, qkv2_w, out2_w, out2_b, res2_w, B, stream);

  // ---- classifier head ----
  k_head<<<dim3(1), 256, 0, stream>>>(B.Hbuf2, normf_g, normf_b, fc2_w, fc2_b, out);
}
